// NetworkBasic_43997644980513
// MI455X (gfx1250) — matrix operand order repaired and checked by reading, not compiled, not run on hardware
//
#include <hip/hip_runtime.h>

typedef __attribute__((ext_vector_type(16))) _Float16 v16h;
typedef __attribute__((ext_vector_type(8)))  float    v8f;

#define NB 4      // batch
#define CC 4      // channels (in == out for both layers)
#define HH 128
#define WW 128
#define TT 100    // timesteps
#define TPAD 112  // 7 x 16 padded time extent
#define TBLK 7    // 16-wide time tiles
#define XB 8      // x positions per workgroup (one per wave)

__device__ __forceinline__ int iclamp(int v, int lo, int hi) {
    return v < lo ? lo : (v > hi ? hi : v);
}

// Compile-time tap -> staged-LDS half-offset. K slots >= KTOT map to the
// zeroed pad region at ROWS*TPAD (safe for any cmn = xl*TPAD + t < XB*TPAD).
template<int KW, int RAD, int KTOT>
constexpr int koff(int k) {
    constexpr int TAPS = KW * KW;
    constexpr int XW   = XB + 2 * RAD;
    constexpr int ROWS = CC * KW * XW;
    if (k >= KTOT) return ROWS * TPAD;
    int ci = k / TAPS, rr = k % TAPS;
    int dy = rr / KW,  dx = rr % KW;
    return ((ci * KW + dy) * XW + dx) * TPAD;
}

// ---------------------------------------------------------------------------
// Per-pixel alpha-PSP scan:  M' = d*(M+Q); y = c*M'; Q' = d*Q + x
// ---------------------------------------------------------------------------
__global__ void psp_scan_kernel(const float* __restrict__ x,
                                _Float16* __restrict__ psp,
                                float d, float c, int npix)
{
    int p = blockIdx.x * blockDim.x + threadIdx.x;
    if (p >= npix) return;
    const float4* xv = reinterpret_cast<const float4*>(x + (size_t)p * TT);
    _Float16* op = psp + (size_t)p * TT;
    float Q = 0.f, M = 0.f;
    for (int tb = 0; tb < TT / 4; ++tb) {
        __builtin_prefetch(xv + tb + 8, 0, 1);   // global_prefetch_b8
        float4 v = xv[tb];
        float in[4] = {v.x, v.y, v.z, v.w};
#pragma unroll
        for (int i = 0; i < 4; ++i) {
            float Mn = d * (M + Q);
            Q = d * Q + in[i];
            M = Mn;
            op[tb * 4 + i] = (_Float16)(c * Mn);
        }
    }
}

// ---------------------------------------------------------------------------
// Fused SLAYER spike (refractory alpha feedback) + next-layer alpha PSP.
// ---------------------------------------------------------------------------
__global__ void spike_psp_kernel(const float* __restrict__ u,
                                 _Float16* __restrict__ psp2,
                                 float dr, float cr, float th,
                                 float d2, float c2, int npix)
{
    int p = blockIdx.x * blockDim.x + threadIdx.x;
    if (p >= npix) return;
    const float4* uv = reinterpret_cast<const float4*>(u + (size_t)p * TT);
    _Float16* op = psp2 + (size_t)p * TT;
    float Qr = 0.f, Mr = 0.f, Q2 = 0.f, M2 = 0.f;
    for (int tb = 0; tb < TT / 4; ++tb) {
        __builtin_prefetch(uv + tb + 8, 0, 1);
        float4 v = uv[tb];
        float in[4] = {v.x, v.y, v.z, v.w};
#pragma unroll
        for (int i = 0; i < 4; ++i) {
            float Mrn  = dr * (Mr + Qr);
            float utot = in[i] + cr * Mrn;
            float s    = (utot >= th) ? 1.f : 0.f;
            Qr = dr * Qr + s;  Mr = Mrn;
            float M2n = d2 * (M2 + Q2);
            op[tb * 4 + i] = (_Float16)(c2 * M2n);
            Q2 = d2 * Q2 + s;  M2 = M2n;
        }
    }
}

// ---------------------------------------------------------------------------
// Final spike layer: u2 -> binary spikes (float32 output).
// ---------------------------------------------------------------------------
__global__ void spike_out_kernel(const float* __restrict__ u,
                                 float* __restrict__ out,
                                 float dr, float cr, float th, int npix)
{
    int p = blockIdx.x * blockDim.x + threadIdx.x;
    if (p >= npix) return;
    const float4* uv = reinterpret_cast<const float4*>(u + (size_t)p * TT);
    float* op = out + (size_t)p * TT;
    float Qr = 0.f, Mr = 0.f;
    for (int tb = 0; tb < TT / 4; ++tb) {
        __builtin_prefetch(uv + tb + 8, 0, 1);
        float4 v = uv[tb];
        float in[4] = {v.x, v.y, v.z, v.w};
#pragma unroll
        for (int i = 0; i < 4; ++i) {
            float Mrn  = dr * (Mr + Qr);
            float utot = in[i] + cr * Mrn;
            float s    = (utot >= th) ? 1.f : 0.f;
            Qr = dr * Qr + s;  Mr = Mrn;
            op[tb * 4 + i] = s;
        }
    }
}

// ---------------------------------------------------------------------------
// LDS-staged implicit-GEMM conv via v_wmma_f32_16x16x32_f16, time as M dim.
//   D[16t x 16o] += A[16t x 32k] x B[32k x 16o],  k = (c,ky,kx) taps.
//
// Block = 8 waves = 8 consecutive x of one (n,y) row. Block stages the haloed
// f16 window [C][KW][XB+2R][TPAD] (zero-padded in space AND time) plus an
// extra zeroed XB*TPAD pad region (targets for K-padding slots) plus a
// zero-padded [16][K] weight tile into LDS with coalesced b32 loads.
//
// Hot loop per A element: v_cndmask(const,const) + v_add + ds_load_u16
// (tap decode folded at compile time via koff); B fragments: 2x ds_load_b128.
//
// Fragment layouts per CDNA5 ISA §7.12.2 (wave32):
//   A (16x32 f16): row M = lane&15; lanes<16 hold K {0..7,16..23}, lanes>=16
//                  hold K {8..15,24..31} (v16h elems 0..15).
//   B (32x16 f16): col N = lane&15; lanes<16 hold K {0..7,16..23}, lanes>=16
//                  hold K {8..15,24..31}.
//   D (16x16 f32): N = lane&15; vgpr r holds M = r (+8 for lanes>=16).
// ---------------------------------------------------------------------------
template<int KW, int RAD, int KTOT, int CHUNKS>
__global__ void __launch_bounds__(256)
conv_wmma_kernel(const _Float16* __restrict__ psp,
                 const float* __restrict__ wgt,
                 float* __restrict__ u)
{
    constexpr int KDIM  = CHUNKS * 32;
    constexpr int XW    = XB + 2 * RAD;         // staged x extent
    constexpr int ROWS  = CC * KW * XW;         // staged (c,dy,x) rows
    constexpr int ZROWS = XB;                   // zero pad rows (K-pad target)
    constexpr int RDW   = TPAD / 2;             // dwords per staged row (56)

    __shared__ uint32_t sAraw[(ROWS + ZROWS) * RDW]; // [c][dy][x][t] + zeros
    __shared__ __align__(16) _Float16 sW[16 * KDIM]; // [o][k], zero-padded
    _Float16* sA = reinterpret_cast<_Float16*>(sAraw);

    // ---- block position -------------------------------------------------
    int bid = blockIdx.x;
    int xb  = bid % (WW / XB);
    int r0  = bid / (WW / XB);
    int y   = r0 % HH;
    int n   = r0 / HH;
    int x0  = xb * XB;

    // ---- stage weights (branchless: clamped address + select) -----------
    for (int i = threadIdx.x; i < 16 * KDIM; i += 256) {
        int o = i / KDIM, k = i % KDIM;
        bool ok  = (o < CC) & (k < KTOT);
        float wv = wgt[iclamp(o, 0, CC - 1) * KTOT + iclamp(k, 0, KTOT - 1)];
        sW[i] = ok ? (_Float16)wv : (_Float16)0.f;
    }

    // ---- stage haloed activation window + zero pad (coalesced b32) ------
    for (int i = threadIdx.x; i < (ROWS + ZROWS) * RDW; i += 256) {
        int row = i / RDW, dw = i % RDW;
        int rc  = iclamp(row, 0, ROWS - 1);
        int c   = rc / (KW * XW);
        int rr  = rc % (KW * XW);
        int dy  = rr / XW, xi = rr % XW;
        int yy  = y + dy - RAD;
        int xx  = x0 + xi - RAD;
        int t0  = dw * 2;
        bool ok = (row < ROWS) &
                  ((unsigned)yy < (unsigned)HH) & ((unsigned)xx < (unsigned)WW) &
                  (t0 < TT);
        size_t gpix = (((size_t)n * CC + c) * HH + iclamp(yy, 0, HH - 1)) * WW +
                      iclamp(xx, 0, WW - 1);
        const uint32_t* src = reinterpret_cast<const uint32_t*>(
            psp + gpix * TT + iclamp(t0, 0, TT - 2));
        uint32_t v = *src;                       // unconditional, clamped addr
        sAraw[i] = ok ? v : 0u;
    }
    __syncthreads();

    // ---- per-wave GEMM ---------------------------------------------------
    const int  lane   = threadIdx.x & 31;
    const int  xl     = threadIdx.x >> 5;        // wave id = x offset in strip
    const int  x      = x0 + xl;
    const int  oN     = lane & 15;               // D column = output channel
    const int  m      = lane & 15;               // A row    = time offset
    const int  khi    = (lane < 16) ? 0 : 8;
    const bool lohalf = (lane < 16);

    union F { v16h v; _Float16 h[16]; uint4 q[2]; };

    // B fragments once per wave: two 8-half runs (K base, base+16) -> 2x ds_load_b128
    const _Float16* bp = &sW[oN * KDIM + (lohalf ? 0 : 8)];
    F bfrag[CHUNKS];
#pragma unroll
    for (int kc = 0; kc < CHUNKS; ++kc) {
        bfrag[kc].q[0] = *reinterpret_cast<const uint4*>(bp + kc * 32);
        bfrag[kc].q[1] = *reinterpret_cast<const uint4*>(bp + kc * 32 + 16);
    }

    float* ubase = u + ((((size_t)n * CC + oN) * HH + y) * WW + x) * TT;

    int cmn = xl * TPAD + m;                     // staged (x,t) base, halves
    for (int tb = 0; tb < TBLK; ++tb, cmn += 16) {
        v8f acc = {0.f, 0.f, 0.f, 0.f, 0.f, 0.f, 0.f, 0.f};
#pragma unroll
        for (int kc = 0; kc < CHUNKS; ++kc) {
            F a;
#pragma unroll
            for (int j = 0; j < 16; ++j) {
                const int klo = kc * 32 + (j < 8 ? j : j + 8); // compile-time
                const int olo = koff<KW, RAD, KTOT>(klo);      // compile-time
                const int ohi = koff<KW, RAD, KTOT>(klo + 8);  // compile-time
                int off = lohalf ? olo : ohi;                  // cndmask
                a.h[j] = sA[cmn + off];                        // ds_load_u16
            }
            acc = __builtin_amdgcn_wmma_f32_16x16x32_f16(
                      false, a.v, false, bfrag[kc].v, (short)0, acc,
                      false, false);
        }
        // store D rows (output channels 0..3; mask tail time tile only)
        if (oN < CC) {
            if (tb < TBLK - 1) {
#pragma unroll
                for (int r2 = 0; r2 < 8; ++r2)
                    ubase[tb * 16 + r2 + khi] = acc[r2];
            } else {
#pragma unroll
                for (int r2 = 0; r2 < 8; ++r2) {
                    int tm = tb * 16 + r2 + khi;
                    if (tm < TT) ubase[tm] = acc[r2];
                }
            }
        }
    }
}

// ---------------------------------------------------------------------------
// Host launcher. Workspace: psp (f16, 52.4 MB) + u (f32, 104.9 MB), ping-pong.
// ---------------------------------------------------------------------------
extern "C" void kernel_launch(void* const* d_in, const int* in_sizes, int n_in,
                              void* d_out, int out_size, void* d_ws, size_t ws_size,
                              hipStream_t stream)
{
    (void)in_sizes; (void)n_in; (void)out_size; (void)ws_size;

    const float* x  = (const float*)d_in[0];
    const float* w1 = (const float*)d_in[1];
    const float* w2 = (const float*)d_in[2];
    float* out = (float*)d_out;

    const int    NPIX  = NB * CC * HH * WW;            // 262144 pixels
    const size_t NELEM = (size_t)NPIX * TT;            // 26,214,400
    const int    NCBLK = NB * HH * (WW / XB);          // 8192 conv blocks

    _Float16* psp  = (_Float16*)d_ws;
    float*    ubuf = (float*)((char*)d_ws + NELEM * sizeof(_Float16));

    // Recurrence constants (Ts = 1).
    const float d1 = 0.36787944117f;   // exp(-1),  tau = 1
    const float c1 = 2.71828182846f;   // e * (1/1)
    const float d2 = 0.60653065971f;   // exp(-0.5), tau = 2
    const float c2 = 1.35914091423f;   // 0.5 * e
    const float cr1 = -81.54845485f;   // -1*30*(1/1)*e   (layer-1 refractory)
    const float cr2 = -67.95704571f;   // -1*50*(1/2)*e   (layer-2 refractory)

    dim3 blk(256);
    dim3 gscan(NPIX / 256);     // 1024
    dim3 gconv(NCBLK);          // 8192

    // 1) psp1 = alpha_psp(x, tau=1)                       x -> psp (f16)
    psp_scan_kernel<<<gscan, blk, 0, stream>>>(x, psp, d1, c1, NPIX);

    // 2) u1 = conv5x5(psp1, w1)                           psp -> ubuf
    conv_wmma_kernel<5, 2, 100, 4><<<gconv, blk, 0, stream>>>(psp, w1, ubuf);

    // 3) s1 = spike(u1, th=30); psp2 = alpha_psp(s1, 2)   ubuf -> psp (reuse)
    spike_psp_kernel<<<gscan, blk, 0, stream>>>(ubuf, psp, d1, cr1, 30.f, d2, c2, NPIX);

    // 4) u2 = conv3x3(psp2, w2)                           psp -> ubuf (reuse)
    conv_wmma_kernel<3, 1, 36, 2><<<gconv, blk, 0, stream>>>(psp, w2, ubuf);

    // 5) s2 = spike(u2, th=50, tauRef=2)                  ubuf -> out
    spike_out_kernel<<<gscan, blk, 0, stream>>>(ubuf, out, d2, cr2, 50.f, NPIX);
}
